// SwinBlock_23416161697841
// MI455X (gfx1250) — matrix operand order repaired and checked by reading, not compiled, not run on hardware
//
#include <hip/hip_runtime.h>
#include <stdint.h>

// ---------------------------------------------------------------------------
// Swin block for MI455X (gfx1250, wave32, WMMA bf16 16x16x32, fp32 accum,
// async global->LDS DMA staging for the GEMM pipeline)
// ---------------------------------------------------------------------------

typedef __bf16 bf16_t;
typedef __attribute__((ext_vector_type(16))) __bf16 v16bf;
typedef __attribute__((ext_vector_type(8)))  __bf16 v8bf;
typedef __attribute__((ext_vector_type(8)))  float  v8f;

#define B_    32
#define H_    56
#define W_    56
#define C_    384
#define NH_   12
#define HD_   32
#define HID_  1536
#define NTOK  (B_ * H_ * W_)        // 100352
#define NWIN  (B_ * 64)             // 2048 windows of 49 tokens
#define QKV_LD (3 * C_)             // 1152
#define ATT_SCALE 0.17677669529663687f   // 32^-0.5

// ---- index maps for window partition / reverse ----------------------------
__device__ __forceinline__ int win_row_of_token(int t) {
  int b = t / 3136, hw = t % 3136;
  int r = hw / 56,  c  = hw % 56;
  int win = b * 64 + (r / 7) * 8 + (c / 7);
  int pos = (r % 7) * 7 + (c % 7);
  return win * 49 + pos;
}
__device__ __forceinline__ int token_of_win_row(int wr) {
  int win = wr / 49, pos = wr % 49;
  int b = win / 64,  wi  = win % 64;
  int r = (wi / 8) * 7 + pos / 7;
  int c = (wi % 8) * 7 + pos % 7;
  return b * 3136 + r * 56 + c;
}

// ---- WMMA fragment helpers (contiguous k order, same for A and B) -----------------
// A-frag (16-bit, 16x32): lane L holds row m=L%16; element i -> K index
//   (L>>4)*16 + i  => 16 contiguous halves at p[0..15].
__device__ __forceinline__ v16bf load_frag_a(const bf16_t* p) {
  v8bf lo = *(const v8bf*)(p);
  v8bf hi = *(const v8bf*)(p + 8);
  return __builtin_shufflevector(lo, hi, 0,1,2,3,4,5,6,7,8,9,10,11,12,13,14,15);
}
// B-frag (16-bit, 32x16): lane L holds col n=L&15; element i -> K index
//   (L>>4)*16 + i  => 16 contiguous halves when data is K-contiguous per n.
__device__ __forceinline__ v16bf load_frag_b(const bf16_t* p) {
  v8bf lo = *(const v8bf*)(p);
  v8bf hi = *(const v8bf*)(p + 8);
  return __builtin_shufflevector(lo, hi, 0,1,2,3,4,5,6,7,8,9,10,11,12,13,14,15);
}
__device__ __forceinline__ v8f wmma_bf16(v16bf a, v16bf b, v8f c) {
  return __builtin_amdgcn_wmma_f32_16x16x32_bf16(false, a, false, b,
                                                 (short)0, c, false, false);
}
__device__ __forceinline__ float gelu_f(float v) {
  return 0.5f * v * (1.0f + erff(v * 0.70710678118654752f));
}

// Async global->LDS copy of one 16B chunk (CDNA5 ASYNCcnt path).
// LDS byte address = low 32 bits of generic pointer (ISA 10.2: LDS aperture
// truncates to addr[31:0]).
__device__ __forceinline__ void async_copy_b128(void* lds_ptr, const void* gptr) {
  uint32_t lds = (uint32_t)(uintptr_t)lds_ptr;
  uint64_t ga  = (uint64_t)(uintptr_t)gptr;
  asm volatile("global_load_async_to_lds_b128 %0, %1, off"
               :: "v"(lds), "v"(ga) : "memory");
}
__device__ __forceinline__ void async_wait_all() {
  asm volatile("s_wait_asynccnt 0" ::: "memory");
}

// ---------------------------------------------------------------------------
// fp32 -> bf16 copy (weights)
// ---------------------------------------------------------------------------
__global__ __launch_bounds__(256) void f2bf_kernel(const float* __restrict__ in,
                                                   bf16_t* __restrict__ out, int n) {
  int i = blockIdx.x * 256 + threadIdx.x;
  if (i < n) out[i] = (bf16_t)in[i];
}

// ---------------------------------------------------------------------------
// LayerNorm: one wave (32 lanes) per token, 12 channels/lane, shfl reduce.
// windowed!=0 -> scatter rows into window-partitioned order.
// ---------------------------------------------------------------------------
__global__ __launch_bounds__(256) void ln_kernel(const float* __restrict__ x,
                                                 const float* __restrict__ g,
                                                 const float* __restrict__ b,
                                                 bf16_t* __restrict__ out,
                                                 int windowed) {
  const int lane = threadIdx.x & 31;
  const int wave = threadIdx.x >> 5;
  const int t = blockIdx.x * 8 + wave;
  if (t >= NTOK) return;
  const float* xp = x + (size_t)t * C_;

  float vals[12];
  float s = 0.f;
#pragma unroll
  for (int j = 0; j < 12; ++j) { vals[j] = xp[lane + j * 32]; s += vals[j]; }
#pragma unroll
  for (int m = 16; m >= 1; m >>= 1) s += __shfl_xor(s, m, 32);
  const float mu = s * (1.0f / 384.0f);

  float s2 = 0.f;
#pragma unroll
  for (int j = 0; j < 12; ++j) { float d = vals[j] - mu; s2 += d * d; }
#pragma unroll
  for (int m = 16; m >= 1; m >>= 1) s2 += __shfl_xor(s2, m, 32);
  const float rinv = rsqrtf(s2 * (1.0f / 384.0f) + 1e-5f);

  const int row = windowed ? win_row_of_token(t) : t;
  bf16_t* op = out + (size_t)row * C_;
#pragma unroll
  for (int j = 0; j < 12; ++j) {
    int c = lane + j * 32;
    op[c] = (bf16_t)((vals[j] - mu) * rinv * g[c] + b[c]);
  }
}

// ---------------------------------------------------------------------------
// Tiled bf16 WMMA GEMM: C[M,N] = A[M,K] * Bw[N,K]^T  (Bw is nn.Linear [out,in])
// Block tile 128x128, BK=32, 256 threads = 8 waves in 2(M) x 4(N); each wave
// owns a 64x32 output patch = 4x2 wmma tiles (8 WMMA per k-step).
// Double-buffered LDS, tiles staged with global_load_async_to_lds_b128 so the
// next tile's DMA overlaps the current tile's WMMAs; one barrier per k-step.
// EPI: 0=store bf16 (qkv), 1=bias+window_reverse+residual->f32 (proj),
//      2=bias+GELU->bf16 (fc1), 3=bias+GELU+residual->f32 (fc2)
// ---------------------------------------------------------------------------
template <int EPI>
__global__ __launch_bounds__(256) void gemm_kernel(
    const bf16_t* __restrict__ A, const bf16_t* __restrict__ Bw,
    const float* __restrict__ bias, const float* __restrict__ resid,
    float* __restrict__ outF, bf16_t* __restrict__ outB,
    int M, int N, int K) {
  __shared__ bf16_t At[2][128][40];   // row stride 80B (16B multiple)
  __shared__ bf16_t Bt[2][128][40];   // staged K-contiguous per output col

  const int tid   = threadIdx.x;
  const int lane  = tid & 31;
  const int wave  = tid >> 5;
  const int waveM = wave & 1;        // 2 M groups of 64 rows
  const int waveN = wave >> 1;       // 4 N groups of 32 cols
  const int blockM = blockIdx.x * 128;
  const int blockN = blockIdx.y * 128;
  const int khalf  = lane >> 4;      // 0/1
  const int ncol   = lane & 15;

  // async DMA one 128x32 A tile + 128x32 B tile into buffer p
  auto stage = [&](int k0, int p) {
#pragma unroll
    for (int j = 0; j < 2; ++j) {
      int c   = tid + j * 256;       // 512 chunks of 8 halves
      int row = c >> 2;
      int col = (c & 3) << 3;
      async_copy_b128(&At[p][row][col],
                      A + (size_t)(blockM + row) * K + k0 + col);
    }
#pragma unroll
    for (int j = 0; j < 2; ++j) {
      int c   = tid + j * 256;
      int row = c >> 2;
      int col = (c & 3) << 3;
      async_copy_b128(&Bt[p][row][col],
                      Bw + (size_t)(blockN + row) * K + k0 + col);
    }
  };

  v8f acc[4][2];
#pragma unroll
  for (int i = 0; i < 4; ++i)
#pragma unroll
    for (int j = 0; j < 2; ++j) acc[i][j] = (v8f){};

  stage(0, 0);
  async_wait_all();
  __syncthreads();

  int p = 0;
  for (int k0 = 0; k0 < K; k0 += 32) {
    if (k0 + 32 < K) stage(k0 + 32, p ^ 1);   // overlap DMA with compute

    v16bf af[4], bfv[2];
#pragma unroll
    for (int mi = 0; mi < 4; ++mi)
      af[mi] = load_frag_a(&At[p][waveM * 64 + mi * 16 + ncol][khalf * 16]);
#pragma unroll
    for (int ni = 0; ni < 2; ++ni)
      bfv[ni] = load_frag_b(&Bt[p][waveN * 32 + ni * 16 + ncol][khalf * 16]);
#pragma unroll
    for (int mi = 0; mi < 4; ++mi)
#pragma unroll
      for (int ni = 0; ni < 2; ++ni)
        acc[mi][ni] = wmma_bf16(af[mi], bfv[ni], acc[mi][ni]);

    async_wait_all();
    __syncthreads();
    p ^= 1;
  }

  // epilogue; C/D layout: VGPR r -> row = r + (lane>=16 ? 8 : 0), col = lane&15
  const int rbase = khalf * 8;
#pragma unroll
  for (int mi = 0; mi < 4; ++mi)
#pragma unroll
    for (int ni = 0; ni < 2; ++ni)
#pragma unroll
      for (int r = 0; r < 8; ++r) {
        int row = blockM + waveM * 64 + mi * 16 + rbase + r;
        int col = blockN + waveN * 32 + ni * 16 + ncol;
        float v = acc[mi][ni][r];
        if constexpr (EPI == 0) {
          outB[(size_t)row * N + col] = (bf16_t)v;
        } else if constexpr (EPI == 1) {
          v += bias[col];
          int t = token_of_win_row(row);
          outF[(size_t)t * C_ + col] = resid[(size_t)t * C_ + col] + v;
        } else if constexpr (EPI == 2) {
          outB[(size_t)row * N + col] = (bf16_t)gelu_f(v + bias[col]);
        } else {
          v = gelu_f(v + bias[col]);
          outF[(size_t)row * C_ + col] = resid[(size_t)row * C_ + col] + v;
        }
      }
}

// ---------------------------------------------------------------------------
// Windowed attention: one wave per (window, head). N=49 padded to 64.
// S = Q K^T * scale (4x4 wmma, K=32), mask cols>=49, softmax rows<49,
// O = P V (4x2 wmma, K=64 in two steps, V zero-padded rows>=49).
// ---------------------------------------------------------------------------
__global__ __launch_bounds__(32) void attn_kernel(const bf16_t* __restrict__ qkv,
                                                  bf16_t* __restrict__ attn_out) {
  __shared__ float  S[64][65];
  __shared__ bf16_t P[64][72];     // row stride 144B (16B multiple)
  __shared__ bf16_t Vt[32][72];    // [channel][token], K-contiguous for B-frags

  const int win  = blockIdx.x;
  const int head = blockIdx.y;
  const int lane = threadIdx.x;
  const int khalf = lane >> 4;
  const int ncol  = lane & 15;

  const bf16_t* qp = qkv + (size_t)win * 49 * QKV_LD + head * HD_;
  const bf16_t* kp = qp + C_;
  const bf16_t* vp = qp + 2 * C_;

  // stage V transposed + zero-padded
  for (int idx = lane; idx < 64 * 32; idx += 32) {
    int r = idx >> 5, c = idx & 31;
    float fv = (r < 49) ? (float)vp[(size_t)r * QKV_LD + c] : 0.0f;
    Vt[c][r] = (bf16_t)fv;
  }

  // ---- S = Q K^T ----
  v16bf aq[4], bk[4];
#pragma unroll
  for (int mi = 0; mi < 4; ++mi)
    aq[mi] = load_frag_a(qp + (size_t)(mi * 16 + ncol) * QKV_LD + khalf * 16);
#pragma unroll
  for (int ni = 0; ni < 4; ++ni)
    bk[ni] = load_frag_b(kp + (size_t)(ni * 16 + ncol) * QKV_LD + khalf * 16);

  const int rbase = khalf * 8;
#pragma unroll
  for (int mi = 0; mi < 4; ++mi)
#pragma unroll
    for (int ni = 0; ni < 4; ++ni) {
      v8f s = wmma_bf16(aq[mi], bk[ni], (v8f){});
      int n = ni * 16 + ncol;
#pragma unroll
      for (int r = 0; r < 8; ++r) {
        int m = mi * 16 + rbase + r;
        S[m][n] = (n < 49) ? s[r] * ATT_SCALE : -1e30f;
      }
    }

  // ---- softmax per row (lane r and r+32); pad rows -> 0 ----
#pragma unroll
  for (int pass = 0; pass < 2; ++pass) {
    int r = lane + pass * 32;
    if (r < 49) {
      float mx = -1e30f;
      for (int c = 0; c < 64; ++c) mx = fmaxf(mx, S[r][c]);
      float sum = 0.f;
      for (int c = 0; c < 64; ++c) { float e = __expf(S[r][c] - mx); sum += e; S[r][c] = e; }
      float inv = 1.0f / sum;
      for (int c = 0; c < 64; ++c) P[r][c] = (bf16_t)(S[r][c] * inv);
    } else {
      for (int c = 0; c < 64; ++c) P[r][c] = (bf16_t)0.0f;
    }
  }

  // ---- O = P V ----
  v8f o[4][2];
#pragma unroll
  for (int mi = 0; mi < 4; ++mi)
#pragma unroll
    for (int ni = 0; ni < 2; ++ni) o[mi][ni] = (v8f){};
#pragma unroll
  for (int ks = 0; ks < 2; ++ks) {
    v16bf ap[4], bv[2];
#pragma unroll
    for (int mi = 0; mi < 4; ++mi)
      ap[mi] = load_frag_a(&P[mi * 16 + ncol][ks * 32 + khalf * 16]);
#pragma unroll
    for (int ni = 0; ni < 2; ++ni)
      bv[ni] = load_frag_b(&Vt[ni * 16 + ncol][ks * 32 + khalf * 16]);
#pragma unroll
    for (int mi = 0; mi < 4; ++mi)
#pragma unroll
      for (int ni = 0; ni < 2; ++ni)
        o[mi][ni] = wmma_bf16(ap[mi], bv[ni], o[mi][ni]);
  }

  // store O rows < 49
#pragma unroll
  for (int mi = 0; mi < 4; ++mi)
#pragma unroll
    for (int ni = 0; ni < 2; ++ni)
#pragma unroll
      for (int r = 0; r < 8; ++r) {
        int m = mi * 16 + rbase + r;
        if (m < 49) {
          int col = head * HD_ + ni * 16 + ncol;
          attn_out[((size_t)win * 49 + m) * C_ + col] = (bf16_t)o[mi][ni][r];
        }
      }
}

// ---------------------------------------------------------------------------
// Driver
// ---------------------------------------------------------------------------
extern "C" void kernel_launch(void* const* d_in, const int* in_sizes, int n_in,
                              void* d_out, int out_size, void* d_ws, size_t ws_size,
                              hipStream_t stream) {
  const float* x      = (const float*)d_in[0];
  const float* ln1_w  = (const float*)d_in[1];
  const float* ln1_b  = (const float*)d_in[2];
  const float* qkv_w  = (const float*)d_in[3];
  const float* proj_w = (const float*)d_in[4];
  const float* proj_b = (const float*)d_in[5];
  const float* ln2_w  = (const float*)d_in[6];
  const float* ln2_b  = (const float*)d_in[7];
  const float* fc1_w  = (const float*)d_in[8];
  const float* fc1_b  = (const float*)d_in[9];
  const float* fc2_w  = (const float*)d_in[10];
  const float* fc2_b  = (const float*)d_in[11];
  float* out = (float*)d_out;

  char* ws = (char*)d_ws;
  size_t off = 0;
  auto take = [&](size_t bytes) -> char* {
    char* p = ws + off;
    off = (off + bytes + 255) & ~(size_t)255;
    return p;
  };
  bf16_t* wqkv  = (bf16_t*)take((size_t)3 * C_ * C_ * 2);
  bf16_t* wproj = (bf16_t*)take((size_t)C_ * C_ * 2);
  bf16_t* wfc1  = (bf16_t*)take((size_t)HID_ * C_ * 2);
  bf16_t* wfc2  = (bf16_t*)take((size_t)C_ * HID_ * 2);
  bf16_t* xw    = (bf16_t*)take((size_t)NTOK * C_ * 2);
  bf16_t* qkv   = (bf16_t*)take((size_t)(NTOK + 32) * QKV_LD * 2);  // +pad rows
  bf16_t* attn  = (bf16_t*)take((size_t)NTOK * C_ * 2);
  float*  x1    = (float*) take((size_t)NTOK * C_ * 4);
  bf16_t* xn2   = (bf16_t*)take((size_t)NTOK * C_ * 2);
  bf16_t* h1    = (bf16_t*)take((size_t)NTOK * HID_ * 2);

  auto cvt = [&](const float* src, bf16_t* dst, int n) {
    f2bf_kernel<<<(n + 255) / 256, 256, 0, stream>>>(src, dst, n);
  };
  cvt(qkv_w,  wqkv,  3 * C_ * C_);
  cvt(proj_w, wproj, C_ * C_);
  cvt(fc1_w,  wfc1,  HID_ * C_);
  cvt(fc2_w,  wfc2,  C_ * HID_);

  // LN1 + window partition -> xw (bf16, windowed row order)
  ln_kernel<<<NTOK / 8, 256, 0, stream>>>(x, ln1_w, ln1_b, xw, 1);

  // QKV GEMM: [100352,384] x [384,1152]
  gemm_kernel<0><<<dim3(NTOK / 128, QKV_LD / 128), 256, 0, stream>>>(
      xw, wqkv, nullptr, nullptr, nullptr, qkv, NTOK, QKV_LD, C_);

  // windowed MHSA
  attn_kernel<<<dim3(NWIN, NH_), 32, 0, stream>>>(qkv, attn);

  // proj + bias + window_reverse + residual -> x1 (fp32, token order)
  gemm_kernel<1><<<dim3(NTOK / 128, C_ / 128), 256, 0, stream>>>(
      attn, wproj, proj_b, x, x1, nullptr, NTOK, C_, C_);

  // LN2 -> xn2 (bf16, token order)
  ln_kernel<<<NTOK / 8, 256, 0, stream>>>(x1, ln2_w, ln2_b, xn2, 0);

  // fc1 + bias + GELU -> h1
  gemm_kernel<2><<<dim3(NTOK / 128, HID_ / 128), 256, 0, stream>>>(
      xn2, wfc1, fc1_b, nullptr, nullptr, h1, NTOK, HID_, C_);

  // fc2 + bias + GELU + residual -> out
  gemm_kernel<3><<<dim3(NTOK / 128, C_ / 128), 256, 0, stream>>>(
      h1, wfc2, fc2_b, x1, out, nullptr, NTOK, C_, HID_);
}
